// NeighborEmbedding_5746666242090
// MI455X (gfx1250) — matrix operand order repaired and checked by reading, not compiled, not run on hardware
//
#include <hip/hip_runtime.h>
#include <hip/hip_bf16.h>

#define BB   32
#define NN   4096
#define KNB  32
#define SS1  512
#define SS2  256
#define EPSBN 1e-5f
#define MAXKD 256

typedef __bf16 bf16;
typedef __attribute__((ext_vector_type(16))) __bf16 v16bf;
typedef __attribute__((ext_vector_type(8)))  float  v8f;

// 16-byte global -> LDS asynchronous copy (CDNA5 async-LDS path, ASYNCcnt-tracked)
__device__ __forceinline__ void async_copy16(const void* g, void* l) {
#if defined(__gfx1250__)
  asm volatile("global_load_async_to_lds_b128 %0, %1, off"
               :: "v"((unsigned int)(unsigned long long)l), "v"(g)
               : "memory");
#else
  *(uint4*)l = *(const uint4*)g;
#endif
}

__device__ __forceinline__ void async_fence() {
#if defined(__gfx1250__)
  asm volatile("s_wait_asynccnt 0" ::: "memory");
  __builtin_amdgcn_s_wait_tensorcnt(0);
#endif
}

// ---------------- embedding layer 0: h[b,n,o] = sum_c x[b,c,n] * w[o,c], C=3, O=64
__global__ void mlp0_kernel(const float* __restrict__ x, const float* __restrict__ w,
                            float* __restrict__ h, int total) {
  int t = blockIdx.x * blockDim.x + threadIdx.x;
  if (t >= total) return;
  int o  = t & 63;
  int bn = t >> 6;
  int b  = bn >> 12;        // N = 4096
  int n  = bn & 4095;
  const float* xb = x + (size_t)b * 3 * NN;
  float s = xb[n] * w[o*3 + 0] + xb[NN + n] * w[o*3 + 1] + xb[2*NN + n] * w[o*3 + 2];
  h[t] = s;
}

// ---------------- per-channel batch stats over R rows (train-mode BN)
__global__ void bn_stats_kernel(const float* __restrict__ h, int R, int C,
                                float* __restrict__ meanv, float* __restrict__ rstdv) {
  __shared__ float ss[256];
  __shared__ float sq[256];
  int c = blockIdx.x, tid = threadIdx.x;
  float s = 0.f, q = 0.f;
  for (int r = tid; r < R; r += 256) {
    float v = h[(size_t)r * C + c];
    s += v; q += v * v;
  }
  ss[tid] = s; sq[tid] = q; __syncthreads();
  for (int st = 128; st > 0; st >>= 1) {
    if (tid < st) { ss[tid] += ss[tid+st]; sq[tid] += sq[tid+st]; }
    __syncthreads();
  }
  if (tid == 0) {
    float mu  = ss[0] / (float)R;
    float var = sq[0] / (float)R - mu * mu;
    meanv[c] = mu;
    rstdv[c] = rsqrtf(fmaxf(var, 0.f) + EPSBN);
  }
}

// ---------------- normalize + relu; emit f32 (for pooling / next stage) and bf16 (WMMA A)
__global__ void bn_relu_kernel(const float* __restrict__ h, int C, int total,
                               const float* __restrict__ meanv, const float* __restrict__ rstdv,
                               const float* __restrict__ g, const float* __restrict__ bta,
                               float* __restrict__ out32, bf16* __restrict__ out16) {
  int t = blockIdx.x * blockDim.x + threadIdx.x;
  if (t >= total) return;
  int c = t % C;
  float y = (h[t] - meanv[c]) * rstdv[c] * g[c] + bta[c];
  y = fmaxf(y, 0.f);
  out32[t] = y;
  out16[t] = (bf16)y;
}

__global__ void cvt_bf16_kernel(const float* __restrict__ in, bf16* __restrict__ out, int n) {
  int t = blockIdx.x * blockDim.x + threadIdx.x;
  if (t < n) out[t] = (bf16)in[t];
}

// ---------------- WMMA GEMM with async-LDS A staging.
// Block = 128 threads (4 waves). Block tile: 16 rows x 64 cols.
// A tile (16 x Kd bf16) loaded Global->LDS once via global_load_async_to_lds_b128,
// shared by all 4 waves; W fragments streamed from L2 (16 contiguous bf16 per lane).
__global__ void __launch_bounds__(128)
wmma_gemm_kernel(const bf16* __restrict__ A, const bf16* __restrict__ W,
                 float* __restrict__ Cout, int Kd, int O) {
  __shared__ bf16 Atile[16 * MAXKD];

  int tid  = threadIdx.x;
  int wave = tid >> 5;
  int lane = tid & 31;
  int m  = lane & 15;
  int hi = lane >> 4;

  size_t row0 = (size_t)blockIdx.x * 16;
  int    col0 = blockIdx.y * 64 + wave * 16;

  // ---- stage A tile into LDS: 16 rows x Kd bf16, 16B chunks, async
  int chunksPerRow = Kd >> 3;               // 16B = 8 bf16
  int nchunks = 16 * chunksPerRow;
  for (int ch = tid; ch < nchunks; ch += 128) {
    int r  = ch / chunksPerRow;
    int o8 = (ch % chunksPerRow) * 8;
    async_copy16(A + (row0 + r) * (size_t)Kd + o8, &Atile[r * Kd + o8]);
  }
  async_fence();          // per-wave: own async copies landed in LDS
  __syncthreads();        // cross-wave visibility

  v8f acc = {};
  const bf16* arow = &Atile[m * Kd];
  const bf16* wrow = W + (size_t)(col0 + m) * Kd + hi * 16;

  for (int k0 = 0; k0 < Kd; k0 += 32) {
    __builtin_prefetch(wrow + k0 + 32, 0, 1);   // global_prefetch_b8 on next W chunk
    v16bf a, bfr;
#pragma unroll
    for (int i = 0; i < 16; ++i) {              // A: 16 contiguous K per lane-half (from LDS)
      int kk = hi * 16 + i;
      a[i] = arow[k0 + kk];
    }
#pragma unroll
    for (int i = 0; i < 16; ++i)                // B: 16 contiguous K per lane-half
      bfr[i] = wrow[k0 + i];
    acc = __builtin_amdgcn_wmma_f32_16x16x32_bf16(false, a, false, bfr,
                                                  (short)0, acc, false, false);
  }
  int n = col0 + m;                             // C/D: lanes 0-15 N=lane, VGPR r -> M=r(+8)
#pragma unroll
  for (int r = 0; r < 8; ++r)
    Cout[(row0 + r + 8 * hi) * (size_t)O + n] = acc[r];
}

// ---------------- farthest point sampling, one block per batch, dmin resident in LDS
__global__ void fps_kernel(const float* __restrict__ src, long bstride, int nstride, int cstride,
                           int Np, int Sp, int* __restrict__ out) {
  __shared__ float dmin[4096];
  __shared__ float sred[256];
  __shared__ int   ired[256];
  int b = blockIdx.x, tid = threadIdx.x;
  const float* sb = src + (size_t)b * bstride;
  for (int n = tid; n < Np; n += 256) dmin[n] = 1e10f;
  __syncthreads();
  int far = 0;
  for (int it = 0; it < Sp; ++it) {
    if (tid == 0) out[b * Sp + it] = far;
    float cx = sb[(size_t)far * nstride];
    float cy = sb[(size_t)far * nstride + cstride];
    float cz = sb[(size_t)far * nstride + 2 * cstride];
    float best = -1.f; int bi = 0;
    for (int n = tid; n < Np; n += 256) {
      float dx = sb[(size_t)n * nstride]               - cx;
      float dy = sb[(size_t)n * nstride + cstride]     - cy;
      float dz = sb[(size_t)n * nstride + 2 * cstride] - cz;
      float d  = dx*dx + dy*dy + dz*dz;
      float mnew = fminf(dmin[n], d);
      dmin[n] = mnew;
      if (mnew > best) { best = mnew; bi = n; }
    }
    sred[tid] = best; ired[tid] = bi;
    __syncthreads();
    for (int s = 128; s > 0; s >>= 1) {
      if (tid < s) {
        float o = sred[tid + s];
        if (o > sred[tid] || (o == sred[tid] && ired[tid + s] < ired[tid])) {
          sred[tid] = o; ired[tid] = ired[tid + s];
        }
      }
      __syncthreads();
    }
    far = ired[0];
    __syncthreads();
  }
}

// ---------------- gather 3D coords of sampled centers
__global__ void gather_xyz_kernel(const float* __restrict__ src, long bstride, int nstride, int cstride,
                                  const int* __restrict__ idx, int Sp,
                                  float* __restrict__ out, int total) {
  int t = blockIdx.x * blockDim.x + threadIdx.x;
  if (t >= total) return;
  int c  = t % 3;
  int bs = t / 3;
  int b  = bs / Sp;
  int s  = bs % Sp;
  int n  = idx[b * Sp + s];
  out[t] = src[(size_t)b * bstride + (size_t)n * nstride + (size_t)c * cstride];
}

// ---------------- exact KNN (K=32 smallest sq-dist) via 32x LDS argmin, one block/center
__global__ void knn_kernel(const float* __restrict__ src, long bstride, int nstride, int cstride,
                           int Np, const float* __restrict__ centers, int Sp,
                           int* __restrict__ knn) {
  __shared__ float dist[4096];
  __shared__ float sred[256];
  __shared__ int   ired[256];
  int bs = blockIdx.x;
  int b  = bs / Sp;
  int tid = threadIdx.x;
  const float* sb  = src + (size_t)b * bstride;
  const float* ctr = centers + (size_t)bs * 3;
  float cx = ctr[0], cy = ctr[1], cz = ctr[2];
  for (int n = tid; n < Np; n += 256) {
    float dx = sb[(size_t)n * nstride]               - cx;
    float dy = sb[(size_t)n * nstride + cstride]     - cy;
    float dz = sb[(size_t)n * nstride + 2 * cstride] - cz;
    dist[n] = dx*dx + dy*dy + dz*dz;
  }
  __syncthreads();
  for (int kk = 0; kk < KNB; ++kk) {
    float best = 3.5e38f; int bi = 0;
    for (int n = tid; n < Np; n += 256) {
      float d = dist[n];
      if (d < best) { best = d; bi = n; }
    }
    sred[tid] = best; ired[tid] = bi;
    __syncthreads();
    for (int s = 128; s > 0; s >>= 1) {
      if (tid < s) {
        float o = sred[tid + s];
        if (o < sred[tid] || (o == sred[tid] && ired[tid + s] < ired[tid])) {
          sred[tid] = o; ired[tid] = ired[tid + s];
        }
      }
      __syncthreads();
    }
    if (tid == 0) { knn[(size_t)bs * KNB + kk] = ired[0]; dist[ired[0]] = 3.5e38f; }
    __syncthreads();
  }
}

// ---------------- build [rel | center] grouped features directly in bf16 (WMMA A operand)
__global__ void build_feat_kernel(const float* __restrict__ f, int Ns, int C,
                                  const int* __restrict__ knn, const int* __restrict__ fps,
                                  int Sp, bf16* __restrict__ out, int total) {
  int t = blockIdx.x * blockDim.x + threadIdx.x;
  if (t >= total) return;
  int C2  = 2 * C;
  int c   = t % C2;
  int row = t / C2;
  int j = row % KNB;
  int s = (row / KNB) % Sp;
  int b = row / (KNB * Sp);
  int ci = fps[b * Sp + s];
  const float* fb = f + (size_t)b * Ns * C;
  float v;
  if (c < C) {
    int ni = knn[(size_t)(b * Sp + s) * KNB + j];
    v = fb[(size_t)ni * C + c] - fb[(size_t)ci * C + c];
  } else {
    v = fb[(size_t)ci * C + (c - C)];
  }
  out[t] = (bf16)v;
}

// ---------------- max over K neighbors; optional [B,S,C] -> [B,C,S] transpose on store
__global__ void maxpool_kernel(const float* __restrict__ h, int Sp, int C,
                               float* __restrict__ out, int transpose, int total) {
  int t = blockIdx.x * blockDim.x + threadIdx.x;
  if (t >= total) return;
  int c = t % C;
  int s = (t / C) % Sp;
  int b = t / (C * Sp);
  size_t base = (size_t)(b * Sp + s) * KNB;
  float m = -3.4e38f;
  for (int j = 0; j < KNB; ++j) m = fmaxf(m, h[(base + j) * (size_t)C + c]);
  if (transpose) out[((size_t)b * C + c) * Sp + s] = m;
  else           out[t] = m;
}

extern "C" void kernel_launch(void* const* d_in, const int* in_sizes, int n_in,
                              void* d_out, int out_size, void* d_ws, size_t ws_size,
                              hipStream_t stream) {
  (void)in_sizes; (void)n_in; (void)out_size; (void)ws_size;
  const float* x    = (const float*)d_in[0];
  const float* w1   = (const float*)d_in[1];
  const float* g1   = (const float*)d_in[2];
  const float* b1   = (const float*)d_in[3];
  const float* w2   = (const float*)d_in[4];
  const float* g2   = (const float*)d_in[5];
  const float* b2   = (const float*)d_in[6];
  const float* s1w1 = (const float*)d_in[7];
  const float* s1g1 = (const float*)d_in[8];
  const float* s1b1 = (const float*)d_in[9];
  const float* s1w2 = (const float*)d_in[10];
  const float* s1g2 = (const float*)d_in[11];
  const float* s1b2 = (const float*)d_in[12];
  const float* s2w1 = (const float*)d_in[13];
  const float* s2g1 = (const float*)d_in[14];
  const float* s2b1 = (const float*)d_in[15];
  const float* s2w2 = (const float*)d_in[16];
  const float* s2g2 = (const float*)d_in[17];
  const float* s2b2 = (const float*)d_in[18];

  char* ws = (char*)d_ws;
  size_t off = 0;
  auto take = [&](size_t bytes) -> char* {
    char* p = ws + off;
    off = (off + bytes + 255) & ~(size_t)255;
    return p;
  };
  const size_t HE = 67108864ull;                 // max rows*channels across stages
  float* h     = (float*)take(HE * 4);           // GEMM outputs (reused per layer)
  bf16*  hb    = (bf16*) take(HE * 2);           // bf16 activations (WMMA A)
  bf16*  featb = (bf16*) take(HE * 2);           // grouped features in bf16
  float* f     = (float*)take((size_t)BB * NN * 64 * 4);    // embedding output
  float* f1    = (float*)take((size_t)BB * SS1 * 128 * 4);  // stage-1 pooled output
  float* meanv = (float*)take(256 * 4);
  float* rstdv = (float*)take(256 * 4);
  int*   fps1  = (int*)  take((size_t)BB * SS1 * 4);
  int*   fps2  = (int*)  take((size_t)BB * SS2 * 4);
  float* nx1   = (float*)take((size_t)BB * SS1 * 3 * 4);
  float* nx2   = (float*)take((size_t)BB * SS2 * 3 * 4);
  int*   knn1  = (int*)  take((size_t)BB * SS1 * KNB * 4);
  int*   knn2  = (int*)  take((size_t)BB * SS2 * KNB * 4);
  bf16*  w2b   = (bf16*) take(64 * 64 * 2);
  bf16*  s1w1b = (bf16*) take(128 * 128 * 2);
  bf16*  s1w2b = (bf16*) take(128 * 128 * 2);
  bf16*  s2w1b = (bf16*) take(256 * 256 * 2);
  bf16*  s2w2b = (bf16*) take(256 * 256 * 2);

  const int R0 = BB * NN;          // 131072
  const int R1 = BB * SS1 * KNB;   // 524288
  const int R2 = BB * SS2 * KNB;   // 262144

  // weights -> bf16
  cvt_bf16_kernel<<<(64*64   + 255)/256, 256, 0, stream>>>(w2,   w2b,   64*64);
  cvt_bf16_kernel<<<(128*128 + 255)/256, 256, 0, stream>>>(s1w1, s1w1b, 128*128);
  cvt_bf16_kernel<<<(128*128 + 255)/256, 256, 0, stream>>>(s1w2, s1w2b, 128*128);
  cvt_bf16_kernel<<<(256*256 + 255)/256, 256, 0, stream>>>(s2w1, s2w1b, 256*256);
  cvt_bf16_kernel<<<(256*256 + 255)/256, 256, 0, stream>>>(s2w2, s2w2b, 256*256);

  // ---- embedding MLP: 3 -> 64 -> 64
  mlp0_kernel<<<(R0*64 + 255)/256, 256, 0, stream>>>(x, w1, h, R0*64);
  bn_stats_kernel<<<64, 256, 0, stream>>>(h, R0, 64, meanv, rstdv);
  bn_relu_kernel<<<(R0*64 + 255)/256, 256, 0, stream>>>(h, 64, R0*64, meanv, rstdv, g1, b1, h, hb);
  wmma_gemm_kernel<<<dim3(R0/16, 1), 128, 0, stream>>>(hb, w2b, h, 64, 64);
  bn_stats_kernel<<<64, 256, 0, stream>>>(h, R0, 64, meanv, rstdv);
  bn_relu_kernel<<<(R0*64 + 255)/256, 256, 0, stream>>>(h, 64, R0*64, meanv, rstdv, g2, b2, f, hb);

  // ---- stage 1: FPS 512, KNN 32, conv 128->128 x2, max over K
  fps_kernel<<<BB, 256, 0, stream>>>(x, (long)3*NN, 1, NN, NN, SS1, fps1);
  gather_xyz_kernel<<<(BB*SS1*3 + 255)/256, 256, 0, stream>>>(x, (long)3*NN, 1, NN, fps1, SS1, nx1, BB*SS1*3);
  knn_kernel<<<BB*SS1, 256, 0, stream>>>(x, (long)3*NN, 1, NN, NN, nx1, SS1, knn1);
  build_feat_kernel<<<(R1*128 + 255)/256, 256, 0, stream>>>(f, NN, 64, knn1, fps1, SS1, featb, R1*128);
  wmma_gemm_kernel<<<dim3(R1/16, 2), 128, 0, stream>>>(featb, s1w1b, h, 128, 128);
  bn_stats_kernel<<<128, 256, 0, stream>>>(h, R1, 128, meanv, rstdv);
  bn_relu_kernel<<<(R1*128 + 255)/256, 256, 0, stream>>>(h, 128, R1*128, meanv, rstdv, s1g1, s1b1, h, hb);
  wmma_gemm_kernel<<<dim3(R1/16, 2), 128, 0, stream>>>(hb, s1w2b, h, 128, 128);
  bn_stats_kernel<<<128, 256, 0, stream>>>(h, R1, 128, meanv, rstdv);
  bn_relu_kernel<<<(R1*128 + 255)/256, 256, 0, stream>>>(h, 128, R1*128, meanv, rstdv, s1g2, s1b2, h, hb);
  maxpool_kernel<<<(BB*SS1*128 + 255)/256, 256, 0, stream>>>(h, SS1, 128, f1, 0, BB*SS1*128);

  // ---- stage 2: FPS 256 (over stage-1 centers), KNN 32, conv 256->256 x2, max, transpose out
  fps_kernel<<<BB, 256, 0, stream>>>(nx1, (long)SS1*3, 3, 1, SS1, SS2, fps2);
  gather_xyz_kernel<<<(BB*SS2*3 + 255)/256, 256, 0, stream>>>(nx1, (long)SS1*3, 3, 1, fps2, SS2, nx2, BB*SS2*3);
  knn_kernel<<<BB*SS2, 256, 0, stream>>>(nx1, (long)SS1*3, 3, 1, SS1, nx2, SS2, knn2);
  build_feat_kernel<<<(R2*256 + 255)/256, 256, 0, stream>>>(f1, SS1, 128, knn2, fps2, SS2, featb, R2*256);
  wmma_gemm_kernel<<<dim3(R2/16, 4), 128, 0, stream>>>(featb, s2w1b, h, 256, 256);
  bn_stats_kernel<<<256, 256, 0, stream>>>(h, R2, 256, meanv, rstdv);
  bn_relu_kernel<<<(R2*256 + 255)/256, 256, 0, stream>>>(h, 256, R2*256, meanv, rstdv, s2g1, s2b1, h, hb);
  wmma_gemm_kernel<<<dim3(R2/16, 4), 128, 0, stream>>>(hb, s2w2b, h, 256, 256);
  bn_stats_kernel<<<256, 256, 0, stream>>>(h, R2, 256, meanv, rstdv);
  bn_relu_kernel<<<(R2*256 + 255)/256, 256, 0, stream>>>(h, 256, R2*256, meanv, rstdv, s2g2, s2b2, h, hb);
  maxpool_kernel<<<(BB*SS2*256 + 255)/256, 256, 0, stream>>>(h, SS2, 256, (float*)d_out, 1, BB*SS2*256);
}
